// ToySymm4bitQLoRALinear_57561151700993
// MI455X (gfx1250) — matrix operand order repaired and checked by reading, not compiled, not run on hardware
//
#include <hip/hip_runtime.h>
#include <stdint.h>

typedef __bf16 bf16_t;
typedef __attribute__((ext_vector_type(16))) __bf16 bf16x16;
typedef __attribute__((ext_vector_type(4)))  __bf16 bf16x4;
typedef __attribute__((ext_vector_type(8)))  float  f32x8;

#define D_IN    4096
#define D_OUT   4096
#define M_TOT   8192
#define R_LORA  16
#define SCALING 2.0f

// CDNA5 GLOBAL_LOAD_ASYNC_TO_LDS_B128 (ASYNCcnt-tracked direct global->LDS DMA)
#define ASYNC_LD_B128(ldsptr, gptr)                                         \
  asm volatile("global_load_async_to_lds_b128 %0, %1, off"                  \
               :: "v"((uint32_t)(uintptr_t)(ldsptr)), "v"(gptr) : "memory")

// partial drain of the in-order async queue: wait until ASYNCcnt <= N
#define WAIT_ASYNC(N) asm volatile("s_wait_asynccnt " #N ::: "memory")

// ---------------------------------------------------------------------------
// Kernel 1: W_eff[o,i] = qweight[o,i]*qscale[o]*meta + SCALING * B[o,:]·A[:,i]
// ---------------------------------------------------------------------------
__global__ __launch_bounds__(256)
void fuse_dequant_lora(const int* __restrict__ qweight,
                       const int* __restrict__ qscale,
                       const float* __restrict__ meta,
                       const float* __restrict__ loraA,
                       const float* __restrict__ loraB,
                       bf16_t* __restrict__ Wb) {
  const int o = blockIdx.x;
  const float scale = (float)qscale[o] * meta[0];
  float Bo[R_LORA];
#pragma unroll
  for (int r = 0; r < R_LORA; ++r) Bo[r] = loraB[o * R_LORA + r] * SCALING;
  for (int i = threadIdx.x; i < D_IN; i += 256) {
    float acc = (float)qweight[(size_t)o * D_IN + i] * scale;
#pragma unroll
    for (int r = 0; r < R_LORA; ++r) acc += Bo[r] * loraA[r * D_IN + i];
    Wb[(size_t)o * D_IN + i] = (bf16_t)acc;
  }
}

// ---------------------------------------------------------------------------
// Kernel 2: fp32 -> bf16 conversion of X
// ---------------------------------------------------------------------------
__global__ __launch_bounds__(256)
void cvt_f32_bf16(const float* __restrict__ src, bf16_t* __restrict__ dst) {
  const size_t idx = ((size_t)blockIdx.x * 256 + threadIdx.x) * 4;
  float4 v = *reinterpret_cast<const float4*>(src + idx);
  bf16x4 o;
  o.x = (bf16_t)v.x; o.y = (bf16_t)v.y; o.z = (bf16_t)v.z; o.w = (bf16_t)v.w;
  *reinterpret_cast<bf16x4*>(dst + idx) = o;
}

// ---------------------------------------------------------------------------
// Kernel 3: C = Xb · Wbᵀ + bias.  128x128 tile, BK=32, TRIPLE-buffered LDS fed
// by async DMA with 2 tiles in flight (s_wait_asynccnt 4 retires only the tile
// about to be consumed). 8 waves × (4x2) v_wmma_f32_16x16x32_bf16.
// ---------------------------------------------------------------------------
#define BM 128
#define BN 128
#define BK 32
#define LDT (BK + 8)          // padded LDS row stride (halves) -> 80B rows
#define NKT (D_IN / BK)       // 128 K-tiles (126 pipelined + 2 drain)

union FragU { uint4 u[2]; bf16x16 v; };

__global__ __launch_bounds__(256)
void wmma_gemm(const bf16_t* __restrict__ Xb, const bf16_t* __restrict__ Wb,
               const float* __restrict__ bias, float* __restrict__ out) {
  __shared__ bf16_t Xs[3][BM * LDT];
  __shared__ bf16_t Ws[3][BN * LDT];

  const int tid  = threadIdx.x;
  const int lane = tid & 31;
  const int wave = tid >> 5;
  const int waveM = (wave >> 2) * 64;   // 2 wave-rows
  const int waveN = (wave & 3) * 32;    // 4 wave-cols
  const int m0 = blockIdx.y * BM;
  const int n0 = blockIdx.x * BN;

  // staging: 512 x 16B chunks per (X,W) tile pair, 2 chunks/thread each
  const int c0 = tid, c1 = tid + 256;
  const int r0 = c0 >> 2, f0 = (c0 & 3) * 8;
  const int r1 = c1 >> 2, f1 = (c1 & 3) * 8;
  const int xo0 = r0 * LDT + f0, xo1 = r1 * LDT + f1;

  // issue pointers advance by BK per tile
  const bf16_t* ax0 = Xb + (size_t)(m0 + r0) * D_IN + f0;
  const bf16_t* ax1 = Xb + (size_t)(m0 + r1) * D_IN + f1;
  const bf16_t* aw0 = Wb + (size_t)(n0 + r0) * D_IN + f0;
  const bf16_t* aw1 = Wb + (size_t)(n0 + r1) * D_IN + f1;

  f32x8 acc[4][2] = {};

  // ISA §7.12.2 fragment addressing (wave32):
  // A 16x32: lane<16 -> K base 0, lane>=16 -> K base 8; second chunk at +16
  // B 32x16: lane<16 -> K base 0, lane>=16 -> K base 8; second chunk at +16
  const int arow = (lane & 15);
  const int akb  = (lane < 16) ? 0 : 8;
  const int bkb  = (lane < 16) ? 0 : 8;

#define ISSUE(BUF)                                                           \
  do {                                                                       \
    ASYNC_LD_B128(&Xs[BUF][xo0], ax0);                                       \
    ASYNC_LD_B128(&Xs[BUF][xo1], ax1);                                       \
    ASYNC_LD_B128(&Ws[BUF][xo0], aw0);                                       \
    ASYNC_LD_B128(&Ws[BUF][xo1], aw1);                                       \
    ax0 += BK; ax1 += BK; aw0 += BK; aw1 += BK;                              \
    __builtin_prefetch(ax0, 0, 1);   /* L2: one tile past async horizon */   \
    __builtin_prefetch(aw0, 0, 1);                                           \
  } while (0)

#define COMPUTE(BUF)                                                         \
  do {                                                                       \
    FragU a_[4], b_[2];                                                      \
    _Pragma("unroll")                                                        \
    for (int ni = 0; ni < 2; ++ni) {                                         \
      const bf16_t* p = &Ws[BUF][(waveN + ni * 16 + arow) * LDT + bkb];      \
      b_[ni].u[0] = *reinterpret_cast<const uint4*>(p);                      \
      b_[ni].u[1] = *reinterpret_cast<const uint4*>(p + 16);                 \
    }                                                                        \
    _Pragma("unroll")                                                        \
    for (int mi = 0; mi < 4; ++mi) {                                         \
      const bf16_t* p = &Xs[BUF][(waveM + mi * 16 + arow) * LDT + akb];      \
      a_[mi].u[0] = *reinterpret_cast<const uint4*>(p);                      \
      a_[mi].u[1] = *reinterpret_cast<const uint4*>(p + 16);                 \
    }                                                                        \
    _Pragma("unroll")                                                        \
    for (int mi = 0; mi < 4; ++mi)                                           \
      _Pragma("unroll")                                                      \
      for (int ni = 0; ni < 2; ++ni)                                         \
        acc[mi][ni] = __builtin_amdgcn_wmma_f32_16x16x32_bf16(               \
            false, a_[mi].v, false, b_[ni].v, (short)0,                      \
            acc[mi][ni], false, false);                                      \
  } while (0)

  // prologue: tiles 0 and 1 in flight; retire tile 0, keep tile 1 running
  ISSUE(0);
  ISSUE(1);
  WAIT_ASYNC(4);
  __syncthreads();

  // software-pipelined main loop: tile i computes on buf i%3 while tiles
  // i+1 (landed/landing) and i+2 (just issued) stream through the async unit.
#pragma unroll 1
  for (int i = 0; i < NKT - 2; i += 3) {
    ISSUE(2); COMPUTE(0); WAIT_ASYNC(4); __syncthreads();
    ISSUE(0); COMPUTE(1); WAIT_ASYNC(4); __syncthreads();
    ISSUE(1); COMPUTE(2); WAIT_ASYNC(4); __syncthreads();
  }
  // drain: tiles NKT-2 (buf 0) and NKT-1 (buf 1)
  COMPUTE(0); WAIT_ASYNC(0); __syncthreads();
  COMPUTE(1);

  // epilogue: C layout (ISA): vgpr r -> M = (lane<16?0:8)+r, lane%16 -> N
#pragma unroll
  for (int ni = 0; ni < 2; ++ni) {
    const int ncol = n0 + waveN + ni * 16 + (lane & 15);
    const float bv = bias[ncol];
#pragma unroll
    for (int mi = 0; mi < 4; ++mi) {
      const int mbase = m0 + waveM + mi * 16 + ((lane < 16) ? 0 : 8);
#pragma unroll
      for (int r = 0; r < 8; ++r)
        out[(size_t)(mbase + r) * D_OUT + ncol] = acc[mi][ni][r] + bv;
    }
  }
#undef ISSUE
#undef COMPUTE
}

// ---------------------------------------------------------------------------
extern "C" void kernel_launch(void* const* d_in, const int* in_sizes, int n_in,
                              void* d_out, int out_size, void* d_ws, size_t ws_size,
                              hipStream_t stream) {
  const float* x       = (const float*)d_in[0];
  const int*   qweight = (const int*)d_in[1];
  const int*   qscale  = (const int*)d_in[2];
  const float* meta    = (const float*)d_in[3];
  const float* bias    = (const float*)d_in[4];
  const float* loraA   = (const float*)d_in[5];
  const float* loraB   = (const float*)d_in[6];
  float* out = (float*)d_out;

  // workspace: Xb (67,108,864 B) then Wb (33,554,432 B)
  bf16_t* Xb = (bf16_t*)d_ws;
  bf16_t* Wb = (bf16_t*)((char*)d_ws + (size_t)M_TOT * D_IN * sizeof(bf16_t));

  fuse_dequant_lora<<<D_OUT, 256, 0, stream>>>(qweight, qscale, meta, loraA, loraB, Wb);

  const size_t nx = (size_t)M_TOT * D_IN;
  cvt_f32_bf16<<<(unsigned)(nx / (256 * 4)), 256, 0, stream>>>(x, Xb);

  dim3 grid(D_OUT / BN, M_TOT / BM);
  wmma_gemm<<<grid, 256, 0, stream>>>(Xb, Wb, bias, out);
}
